// Attention_17179869826
// MI455X (gfx1250) — compile-verified, matrix operand order checked by reading, not run on hardware
//
#include <hip/hip_runtime.h>

// ---------------- problem constants (fixed by the reference) ----------------
constexpr int D_  = 512;   // embed dim
constexpr int H_  = 8;     // heads
constexpr int HD_ = 64;    // head dim
constexpr int B_  = 32;    // graphs
constexpr int L_  = 512;   // padded length

// ---------------- CDNA5 WMMA types ----------------
typedef __bf16  v16bf __attribute__((ext_vector_type(16)));
typedef float   v8f   __attribute__((ext_vector_type(8)));
typedef unsigned int uint4v __attribute__((ext_vector_type(4)));

// fp32 -> bf16, round-to-nearest-even
static __device__ __forceinline__ unsigned short f2bf(float f) {
    unsigned int u = __builtin_bit_cast(unsigned int, f);
    unsigned int r = u + 0x7FFFu + ((u >> 16) & 1u);
    return (unsigned short)(r >> 16);
}

// Load a 16x32 bf16 WMMA fragment (A- or B-operand).
// Per the CDNA5 ISA 16-bit A 16x32 layout:
//   lanes 0-15 : row=lane,    K 0..7 (VGPR0-3) and K 16..23 (VGPR4-7)
//   lanes16-31 : row=lane-16, K 8..15         and K 24..31
// Caller passes p = base + row*ld + k0 + (lane>=16 ? 8 : 0); we fetch p and p+16.
static __device__ __forceinline__ v16bf frag_from(const unsigned short* p) {
    union { struct { uint4v a, b; } s; v16bf v; } u;
    u.s.a = *(const uint4v*)(p);
    u.s.b = *(const uint4v*)(p + 16);
    return u.v;
}

// ---------------- tiny utility kernels ----------------
__global__ void fill_zero_kernel(unsigned int* p, size_t n) {
    size_t t = (size_t)blockIdx.x * blockDim.x + threadIdx.x;
    if (t < n) p[t] = 0u;
}

// dst_bf16[t] = bf16(a[t] + (b? b[t] : 0))
__global__ void cvt_bf16_kernel(const float* __restrict__ a, const float* __restrict__ b,
                                unsigned short* __restrict__ dst, size_t n) {
    size_t t = (size_t)blockIdx.x * blockDim.x + threadIdx.x;
    if (t >= n) return;
    float v = a[t];
    if (b) v += b[t];
    dst[t] = f2bf(v);
}

// Wt[n][k] = bf16(W[k][n])   (W is [K, Nout] row-major; Wt is [Nout, K])
__global__ void wprep_kernel(const float* __restrict__ W, unsigned short* __restrict__ Wt,
                             int K, int Nout) {
    int t = blockIdx.x * blockDim.x + threadIdx.x;
    if (t >= K * Nout) return;
    int n = t / K, k = t - n * K;
    Wt[t] = f2bf(W[(size_t)k * Nout + n]);
}

// GIN neighbor sum: agg[dst] += h[src] over all edges (random dst -> atomics)
__global__ void gin_agg_kernel(const float* __restrict__ h, const int* __restrict__ ei,
                               float* __restrict__ agg, int E) {
    int e = blockIdx.x;
    int d = blockIdx.y * blockDim.x + threadIdx.x;
    int s  = ei[e];
    int dd = ei[E + e];
    atomicAdd(&agg[(size_t)dd * D_ + d], h[(size_t)s * D_ + d]);
}

// ---------------- batch-norm (batch statistics over N nodes) ----------------
__global__ void bn_stats_kernel(const float* __restrict__ h, float* __restrict__ stats, int N) {
    int c = blockIdx.x * blockDim.x + threadIdx.x;   // channel
    float s = 0.f, s2 = 0.f;
    for (int n = blockIdx.y; n < N; n += gridDim.y) {
        float v = h[(size_t)n * D_ + c];
        s += v; s2 += v * v;
    }
    atomicAdd(&stats[c], s);
    atomicAdd(&stats[D_ + c], s2);
}

__global__ void bn_apply_kernel(const float* __restrict__ h, const float* __restrict__ stats,
                                const float* __restrict__ gamma, const float* __restrict__ beta,
                                unsigned short* __restrict__ dst, size_t n, float invN) {
    size_t t = (size_t)blockIdx.x * blockDim.x + threadIdx.x;
    if (t >= n) return;
    int d = (int)(t % D_);
    float mean = stats[d] * invN;
    float var  = stats[D_ + d] * invN - mean * mean;
    float v = (h[t] - mean) * rsqrtf(var + 1e-5f) * gamma[d] + beta[d];
    dst[t] = f2bf(v);
}

// ---------------- bf16 WMMA GEMM: C[M,Nn] = A[M,K] @ Wt[Nn,K]^T + bias ----------------
// 8 waves/block; wave w owns rows [blk*128 + 16w, +16), 64 columns. K-step = 32.
__global__ __launch_bounds__(256) void gemm_bf16_kernel(
    const unsigned short* __restrict__ A, const unsigned short* __restrict__ Wt,
    const float* __restrict__ bias, float* __restrict__ Cf, unsigned short* __restrict__ Cb,
    int M, int Nn, int K, int relu)
{
    int lane = threadIdx.x & 31, w = threadIdx.x >> 5;
    int row0 = blockIdx.x * 128 + w * 16;
    int col0 = blockIdx.y * 64;
    if (row0 >= M) return;
    int hf = lane >> 4, r16 = lane & 15;

    v8f acc0 = {0,0,0,0,0,0,0,0}, acc1 = acc0, acc2 = acc0, acc3 = acc0;

    const unsigned short* arow = A + (size_t)(row0 + r16) * K + hf * 8;
    const unsigned short* b0r  = Wt + (size_t)(col0 +  0 + r16) * K + hf * 8;
    const unsigned short* b1r  = Wt + (size_t)(col0 + 16 + r16) * K + hf * 8;
    const unsigned short* b2r  = Wt + (size_t)(col0 + 32 + r16) * K + hf * 8;
    const unsigned short* b3r  = Wt + (size_t)(col0 + 48 + r16) * K + hf * 8;

    for (int k0 = 0; k0 < K; k0 += 32) {
        __builtin_prefetch(arow + k0 + 32, 0, 1);   // gfx1250 global_prefetch path
        v16bf a = frag_from(arow + k0);
        v16bf b0 = frag_from(b0r + k0);
        acc0 = __builtin_amdgcn_wmma_f32_16x16x32_bf16(false, a, false, b0, (short)0, acc0, false, false);
        v16bf b1 = frag_from(b1r + k0);
        acc1 = __builtin_amdgcn_wmma_f32_16x16x32_bf16(false, a, false, b1, (short)0, acc1, false, false);
        v16bf b2 = frag_from(b2r + k0);
        acc2 = __builtin_amdgcn_wmma_f32_16x16x32_bf16(false, a, false, b2, (short)0, acc2, false, false);
        v16bf b3 = frag_from(b3r + k0);
        acc3 = __builtin_amdgcn_wmma_f32_16x16x32_bf16(false, a, false, b3, (short)0, acc3, false, false);
    }

    v8f* accs[4] = { &acc0, &acc1, &acc2, &acc3 };
    #pragma unroll
    for (int t = 0; t < 4; ++t) {
        int col = col0 + t * 16 + r16;
        float bs = bias[col];
        #pragma unroll
        for (int r = 0; r < 8; ++r) {
            int row = row0 + r + 8 * hf;    // C layout: VGPR r -> M=r (+8 for upper half-wave)
            float v = (*accs[t])[r] + bs;
            if (relu) v = fmaxf(v, 0.f);
            if (Cf) Cf[(size_t)row * Nn + col] = v;
            else    Cb[(size_t)row * Nn + col] = f2bf(v);
        }
    }
}

// ---------------- pad ragged batch; build Qp/Kp [b,h,i,hd] and Vt [b,h,hd,j] in bf16 ----------------
__global__ void pad_kernel(const unsigned short* __restrict__ qkbf, const float* __restrict__ vbuf,
                           const int* __restrict__ ptr,
                           unsigned short* __restrict__ Qp, unsigned short* __restrict__ Kp,
                           unsigned short* __restrict__ Vt, int N) {
    size_t t = (size_t)blockIdx.x * blockDim.x + threadIdx.x;
    if (t >= (size_t)N * D_) return;
    int n = (int)(t / D_), d = (int)(t % D_);
    int lo = 0, hi = B_;
    while (hi - lo > 1) { int mid = (lo + hi) >> 1; if (ptr[mid] <= n) lo = mid; else hi = mid; }
    int b = lo, i = n - ptr[b];
    int h = d >> 6, dh = d & 63;
    size_t qidx = (((size_t)b * H_ + h) * L_ + i) * HD_ + dh;
    Kp[qidx] = qkbf[(size_t)n * (2 * D_) + d];          // k = first half
    Qp[qidx] = qkbf[(size_t)n * (2 * D_) + D_ + d];     // q = second half
    Vt[(((size_t)b * H_ + h) * HD_ + dh) * L_ + i] = f2bf(vbuf[(size_t)n * D_ + d]);
}

// ---------------- flash attention, 1 wave = 16 query rows, WMMA QK^T and P.V ----------------
__global__ __launch_bounds__(256) void attn_kernel(
    const unsigned short* __restrict__ Qp, const unsigned short* __restrict__ Kp,
    const unsigned short* __restrict__ Vt, const unsigned char* __restrict__ mdag,
    const int* __restrict__ ptr, float* __restrict__ AO, float scale)
{
    __shared__ __align__(16) unsigned short plds[8][16 * 32];  // per-wave P tile (bf16)
    int lane = threadIdx.x & 31, w = threadIdx.x >> 5;
    int b = blockIdx.z, hh = blockIdx.y;
    int i0 = (blockIdx.x * 8 + w) * 16;
    int p0 = ptr[b], sz = ptr[b + 1] - p0;
    int hf = lane >> 4, c16 = lane & 15;

    const unsigned short* Qb = Qp + ((size_t)b * H_ + hh) * L_ * HD_;
    const unsigned short* Kb = Kp + ((size_t)b * H_ + hh) * L_ * HD_;
    const unsigned short* Vb = Vt + ((size_t)b * H_ + hh) * HD_ * L_;
    const unsigned char*  Mb = mdag + (size_t)b * L_ * L_;

    // Q fragments (K-dim = hd = 64 -> two 16x32 steps), A-operand layout
    v16bf qf0 = frag_from(Qb + (size_t)(i0 + c16) * HD_ +  0 + hf * 8);
    v16bf qf1 = frag_from(Qb + (size_t)(i0 + c16) * HD_ + 32 + hf * 8);

    v8f o0 = {0,0,0,0,0,0,0,0}, o1 = o0, o2 = o0, o3 = o0;
    float rmax[8], rsum[8];
    #pragma unroll
    for (int r = 0; r < 8; ++r) { rmax[r] = -3.0e38f; rsum[r] = 0.f; }

    int nj = (sz + 31) >> 5;
    for (int jt = 0; jt < nj; ++jt) {
        int j0 = jt * 32;
        v8f s[2];
        #pragma unroll
        for (int u = 0; u < 2; ++u) {
            v8f c = {0,0,0,0,0,0,0,0};
            v16bf kf0 = frag_from(Kb + (size_t)(j0 + u * 16 + c16) * HD_ +  0 + hf * 8);
            c = __builtin_amdgcn_wmma_f32_16x16x32_bf16(false, qf0, false, kf0, (short)0, c, false, false);
            v16bf kf1 = frag_from(Kb + (size_t)(j0 + u * 16 + c16) * HD_ + 32 + hf * 8);
            c = __builtin_amdgcn_wmma_f32_16x16x32_bf16(false, qf1, false, kf1, (short)0, c, false, false);
            s[u] = c;
        }
        // scale + DAG mask + key-pad mask
        #pragma unroll
        for (int u = 0; u < 2; ++u) {
            int j = j0 + u * 16 + c16;
            #pragma unroll
            for (int r = 0; r < 8; ++r) {
                int i = i0 + r + 8 * hf;
                float v = s[u][r] * scale;
                bool m = (j >= sz) || (Mb[(size_t)i * L_ + j] != 0);
                s[u][r] = m ? -1.0e30f : v;
            }
        }
        // online softmax (row = 16 lanes within a half-wave)
        #pragma unroll
        for (int r = 0; r < 8; ++r) {
            float mx = fmaxf(s[0][r], s[1][r]);
            #pragma unroll
            for (int dd = 8; dd >= 1; dd >>= 1) mx = fmaxf(mx, __shfl_xor(mx, dd, 16));
            float mnew = fmaxf(rmax[r], mx);
            float alpha = __expf(rmax[r] - mnew);
            float pa = __expf(s[0][r] - mnew);
            float pb = __expf(s[1][r] - mnew);
            s[0][r] = pa; s[1][r] = pb;
            float ls = pa + pb;
            #pragma unroll
            for (int dd = 8; dd >= 1; dd >>= 1) ls += __shfl_xor(ls, dd, 16);
            rsum[r] = rsum[r] * alpha + ls;
            rmax[r] = mnew;
            o0[r] *= alpha; o1[r] *= alpha; o2[r] *= alpha; o3[r] *= alpha;
        }
        // P (C-layout, fp32) -> LDS bf16 [16 rows][32 keys] -> A-fragment relayout
        unsigned short* lp = plds[w];
        #pragma unroll
        for (int u = 0; u < 2; ++u)
            #pragma unroll
            for (int r = 0; r < 8; ++r)
                lp[(r + 8 * hf) * 32 + u * 16 + c16] = f2bf(s[u][r]);
        asm volatile("s_wait_dscnt 0" ::: "memory");     // CDNA5 split wait: LDS writes visible
        v16bf pf = frag_from(lp + c16 * 32 + hf * 8);
        // P[16x32] @ V[32 keys x 64 hd]  (Vt rows = hd channels, contiguous over keys)
        v16bf vf0 = frag_from(Vb + (size_t)( 0 + c16) * L_ + j0 + hf * 8);
        o0 = __builtin_amdgcn_wmma_f32_16x16x32_bf16(false, pf, false, vf0, (short)0, o0, false, false);
        v16bf vf1 = frag_from(Vb + (size_t)(16 + c16) * L_ + j0 + hf * 8);
        o1 = __builtin_amdgcn_wmma_f32_16x16x32_bf16(false, pf, false, vf1, (short)0, o1, false, false);
        v16bf vf2 = frag_from(Vb + (size_t)(32 + c16) * L_ + j0 + hf * 8);
        o2 = __builtin_amdgcn_wmma_f32_16x16x32_bf16(false, pf, false, vf2, (short)0, o2, false, false);
        v16bf vf3 = frag_from(Vb + (size_t)(48 + c16) * L_ + j0 + hf * 8);
        o3 = __builtin_amdgcn_wmma_f32_16x16x32_bf16(false, pf, false, vf3, (short)0, o3, false, false);
    }
    // normalize and un-pad directly into [N, D] (D index = h*64 + hd)
    #pragma unroll
    for (int r = 0; r < 8; ++r) {
        int i = i0 + r + 8 * hf;
        if (i < sz) {
            float inv = 1.0f / rsum[r];
            size_t base = (size_t)(p0 + i) * D_ + hh * HD_;
            AO[base +  0 + c16] = o0[r] * inv;
            AO[base + 16 + c16] = o1[r] * inv;
            AO[base + 32 + c16] = o2[r] * inv;
            AO[base + 48 + c16] = o3[r] * inv;
        }
    }
}

// ---------------- host orchestration ----------------
extern "C" void kernel_launch(void* const* d_in, const int* in_sizes, int n_in,
                              void* d_out, int out_size, void* d_ws, size_t ws_size,
                              hipStream_t stream) {
    const int N = in_sizes[0] / D_;     // 12160 nodes
    const int E = in_sizes[1] / 2;      // 97280 edges

    const float*         x     = (const float*)d_in[0];
    const int*           ei    = (const int*)d_in[1];
    const unsigned char* mdag  = (const unsigned char*)d_in[2];
    const int*           ptr   = (const int*)d_in[3];
    const float* gw1 = (const float*)d_in[4];
    const float* gb1 = (const float*)d_in[5];
    const float* gw2 = (const float*)d_in[6];
    const float* gb2 = (const float*)d_in[7];
    const float* gamma = (const float*)d_in[8];
    const float* beta  = (const float*)d_in[9];
    const float* sew = (const float*)d_in[10];
    const float* seb = (const float*)d_in[11];
    const float* qkw = (const float*)d_in[12];
    const float* qkb = (const float*)d_in[13];
    const float* vw  = (const float*)d_in[14];
    const float* vb  = (const float*)d_in[15];
    const float* ow  = (const float*)d_in[16];
    const float* ob  = (const float*)d_in[17];
    float* out = (float*)d_out;

    // workspace carving (all offsets 256B aligned)
    char* ws = (char*)d_ws;
    const size_t fND  = (size_t)N * D_ * sizeof(float);          // 24,903,680
    const size_t uND  = (size_t)N * D_ * sizeof(unsigned short); // 12,451,840
    const size_t uPad = (size_t)B_ * H_ * L_ * HD_ * sizeof(unsigned short); // 16,777,216
    size_t cur = 0;
    unsigned short* wbf  = (unsigned short*)(ws + cur); cur += (size_t)11 * D_ * D_ * 2;
    float*          hbuf = (float*)(ws + cur);          cur += fND;   // also reused as attn-out
    float*          agg  = (float*)(ws + cur);          cur += fND;   // also reused as qk bf16 [N,2D]
    float*          vbuf = (float*)(ws + cur);          cur += fND;
    unsigned short* bufA = (unsigned short*)(ws + cur); cur += uND;
    unsigned short* bufZ = (unsigned short*)(ws + cur); cur += uND;
    unsigned short* Qp   = (unsigned short*)(ws + cur); cur += uPad;
    unsigned short* Kp   = (unsigned short*)(ws + cur); cur += uPad;
    unsigned short* Vt   = (unsigned short*)(ws + cur); cur += uPad;
    float*          stats= (float*)(ws + cur);          cur += 2 * D_ * sizeof(float);
    unsigned short* qkbf = (unsigned short*)agg;        // reuse: agg dead after GIN loop
    float*          aobuf= hbuf;                        // reuse: h dead after BN apply

    const dim3 blk(256);
    const size_t cntND = (size_t)N * D_;
    const int cvB = (int)((cntND + 255) / 256);
    const dim3 ggrid((N + 127) / 128, D_ / 64);
    const dim3 ggrid2((N + 127) / 128, (2 * D_) / 64);

    // 1) weights -> bf16, transposed to [out, in]
    const int wpB1 = (D_ * D_ + 255) / 256;
    const int wpB2 = (2 * D_ * D_ + 255) / 256;
    for (int i = 0; i < 3; ++i) {
        wprep_kernel<<<wpB1, blk, 0, stream>>>(gw1 + (size_t)i * D_ * D_, wbf + (size_t)i * D_ * D_, D_, D_);
        wprep_kernel<<<wpB1, blk, 0, stream>>>(gw2 + (size_t)i * D_ * D_, wbf + (size_t)(3 + i) * D_ * D_, D_, D_);
    }
    wprep_kernel<<<wpB1, blk, 0, stream>>>(sew, wbf + (size_t)6 * D_ * D_, D_, D_);
    wprep_kernel<<<wpB2, blk, 0, stream>>>(qkw, wbf + (size_t)7 * D_ * D_, D_, 2 * D_);
    wprep_kernel<<<wpB1, blk, 0, stream>>>(vw,  wbf + (size_t)9 * D_ * D_, D_, D_);
    wprep_kernel<<<wpB1, blk, 0, stream>>>(ow,  wbf + (size_t)10 * D_ * D_, D_, D_);

    // 2) v = x @ v_w + v_b
    cvt_bf16_kernel<<<cvB, blk, 0, stream>>>(x, nullptr, bufA, cntND);
    gemm_bf16_kernel<<<ggrid, blk, 0, stream>>>(bufA, wbf + (size_t)9 * D_ * D_, vb, vbuf, nullptr, N, D_, D_, 0);

    // 3) 3x GIN: agg, h+agg, MLP(relu(W1)+W2), relu
    for (int i = 0; i < 3; ++i) {
        const float* hcur = (i == 0) ? x : hbuf;
        fill_zero_kernel<<<cvB, blk, 0, stream>>>((unsigned int*)agg, cntND);
        gin_agg_kernel<<<dim3(E, D_ / 256), blk, 0, stream>>>(hcur, ei, agg, E);
        cvt_bf16_kernel<<<cvB, blk, 0, stream>>>(hcur, agg, bufA, cntND);
        gemm_bf16_kernel<<<ggrid, blk, 0, stream>>>(bufA, wbf + (size_t)i * D_ * D_, gb1 + (size_t)i * D_,
                                                    nullptr, bufZ, N, D_, D_, 1);
        gemm_bf16_kernel<<<ggrid, blk, 0, stream>>>(bufZ, wbf + (size_t)(3 + i) * D_ * D_, gb2 + (size_t)i * D_,
                                                    hbuf, nullptr, N, D_, D_, 1);
    }

    // 4) batch-norm (batch stats) + se_out linear
    fill_zero_kernel<<<(2 * D_ + 255) / 256, blk, 0, stream>>>((unsigned int*)stats, 2 * D_);
    bn_stats_kernel<<<dim3(D_ / 256, 64), blk, 0, stream>>>(hbuf, stats, N);
    bn_apply_kernel<<<cvB, blk, 0, stream>>>(hbuf, stats, gamma, beta, bufA, cntND, 1.0f / (float)N);
    gemm_bf16_kernel<<<ggrid, blk, 0, stream>>>(bufA, wbf + (size_t)6 * D_ * D_, seb, nullptr, bufZ, N, D_, D_, 0);

    // 5) qk = x_struct @ qk_w + qk_b  (bf16 out, [N, 2D])
    gemm_bf16_kernel<<<ggrid2, blk, 0, stream>>>(bufZ, wbf + (size_t)7 * D_ * D_, qkb, nullptr, qkbf, N, 2 * D_, D_, 0);

    // 6) zero + pad: Qp/Kp [b,h,i,hd], Vt [b,h,hd,j]
    {
        size_t padW = 3 * uPad / 4;  // u32 words across the contiguous Qp|Kp|Vt region
        fill_zero_kernel<<<(int)((padW + 255) / 256), blk, 0, stream>>>((unsigned int*)Qp, padW);
    }
    pad_kernel<<<cvB, blk, 0, stream>>>(qkbf, vbuf, ptr, Qp, Kp, Vt, N);

    // 7) masked flash attention -> [N, D]
    attn_kernel<<<dim3(L_ / 128, H_, B_), blk, 0, stream>>>(Qp, Kp, Vt, mdag, ptr, aobuf, 0.125f);

    // 8) out = attn @ out_w + out_b
    cvt_bf16_kernel<<<cvB, blk, 0, stream>>>(aobuf, nullptr, bufA, cntND);
    gemm_bf16_kernel<<<ggrid, blk, 0, stream>>>(bufA, wbf + (size_t)10 * D_ * D_, ob, out, nullptr, N, D_, D_, 0);

    (void)n_in; (void)out_size; (void)ws_size;
}
